// AttentionSubBlock_60464549593335
// MI455X (gfx1250) — matrix operand order repaired and checked by reading, not compiled, not run on hardware
//
#include <hip/hip_runtime.h>

// ---------------------------------------------------------------- types
typedef __attribute__((ext_vector_type(16))) _Float16 v16h;
typedef __attribute__((ext_vector_type(8)))  _Float16 v8h;
typedef __attribute__((ext_vector_type(8)))  float    v8f;

// ---------------------------------------------------------------- shapes
#define DIM    768
#define HEADS  12
#define HD     64
#define BATCH  4
#define TT     8
#define HH     14
#define WW     14
#define NQ     1568          // TT*HH*WW
#define NKV    392           // 8*7*7
#define NKP    400           // NKV padded to 16
#define NKP2   416           // NKV padded to 32
#define ROWS   (BATCH*NQ)    // 6272
#define QKVCOL (3*DIM)       // 2304
#define EPSLN  1e-6f

// ---------------------------------------------------------------- wmma helpers
__device__ __forceinline__ v16h cat8(v8h lo, v8h hi) {
  return __builtin_shufflevector(lo, hi, 0,1,2,3,4,5,6,7,8,9,10,11,12,13,14,15);
}
// A-matrix 16x32 f16 fragment: lane's halfs are
//   elements 0..15 -> K = kbase + hf*16 + (0..15)       (hf = lane>>4)
__device__ __forceinline__ v16h fragA(const _Float16* row, int kbase, int hf) {
  v8h lo = *(const v8h*)(row + kbase + hf * 16);
  v8h hi = *(const v8h*)(row + kbase + hf * 16 + 8);
  return cat8(lo, hi);
}
// B-matrix 32x16 f16 fragment: lane holds col N=lane&15, contiguous K range
// starting at (lane>>4)*16 — caller passes the already-offset pointer.
__device__ __forceinline__ v16h fragB(const _Float16* p) {
  v8h lo = *(const v8h*)(p);
  v8h hi = *(const v8h*)(p + 8);
  return cat8(lo, hi);
}
__device__ __forceinline__ v8f wmma32(v16h a, v16h b, v8f c) {
  return __builtin_amdgcn_wmma_f32_16x16x32_f16(false, a, false, b, (short)0, c,
                                                false, false);
}
__device__ __forceinline__ float wave_sum(float v) {
  for (int off = 16; off > 0; off >>= 1) v += __shfl_xor(v, off, 32);
  return v;
}
__device__ __forceinline__ float wave_max(float v) {
  for (int off = 16; off > 0; off >>= 1) v = fmaxf(v, __shfl_xor(v, off, 32));
  return v;
}

// CDNA5 async global->LDS copy (ASYNCcnt-tracked DMA, 16 bytes per lane).
// VDST VGPR carries the wave-relative LDS byte offset (flat-pointer low 32
// bits per the LDS aperture rule); address pair in VGPRs, GV mode.
__device__ __forceinline__ void async_copy_b128(const void* g, void* l) {
  unsigned lds_off = (unsigned)(uintptr_t)l;
  asm volatile("global_load_async_to_lds_b128 %0, %1, off"
               :: "v"(lds_off), "v"(g) : "memory");
}
#define WAIT_ASYNCCNT(n) asm volatile("s_wait_asynccnt " #n ::: "memory")

// ---------------------------------------------------------------- K0: LayerNorm(x) -> f16
__global__ __launch_bounds__(256) void ln_cast_kernel(
    const float* __restrict__ x, const float* __restrict__ w,
    const float* __restrict__ b, _Float16* __restrict__ xn) {
  __shared__ float red[256];
  const int row = blockIdx.x, tid = threadIdx.x;
  const float* xr = x + (size_t)row * DIM;
  float v0 = xr[tid], v1 = xr[tid + 256], v2 = xr[tid + 512];
  red[tid] = v0 + v1 + v2;
  __syncthreads();
  for (int o = 128; o > 0; o >>= 1) { if (tid < o) red[tid] += red[tid + o]; __syncthreads(); }
  const float mean = red[0] * (1.0f / DIM);
  __syncthreads();
  const float d0 = v0 - mean, d1 = v1 - mean, d2 = v2 - mean;
  red[tid] = d0 * d0 + d1 * d1 + d2 * d2;
  __syncthreads();
  for (int o = 128; o > 0; o >>= 1) { if (tid < o) red[tid] += red[tid + o]; __syncthreads(); }
  const float rstd = rsqrtf(red[0] * (1.0f / DIM) + EPSLN);
  _Float16* out = xn + (size_t)row * DIM;
  out[tid]       = (_Float16)(d0 * rstd * w[tid]       + b[tid]);
  out[tid + 256] = (_Float16)(d1 * rstd * w[tid + 256] + b[tid + 256]);
  out[tid + 512] = (_Float16)(d2 * rstd * w[tid + 512] + b[tid + 512]);
}

// ---------------------------------------------------------------- weight transpose-cast (KxN f32 -> NxK f16)
__global__ __launch_bounds__(256) void tcast_kernel(
    const float* __restrict__ W, _Float16* __restrict__ Wt, int K, int N) {
  int i = blockIdx.x * 256 + threadIdx.x;
  if (i >= K * N) return;
  int k = i / N, n = i % N;
  Wt[(size_t)n * K + k] = (_Float16)W[(size_t)k * N + n];
}

__global__ __launch_bounds__(256) void zero_h_kernel(_Float16* p, int n) {
  int i = blockIdx.x * 256 + threadIdx.x;
  if (i < n) p[i] = (_Float16)0.0f;
}

// ---------------------------------------------------------------- WMMA GEMM: C(MxN) = A(MxK,f16) * Bt(NxK,f16)^T + bias
// block tile 64x128, 8 waves (4x2), each wave 16x64 via 4 accumulators.
// K-slabs staged via double-buffered async global->LDS DMA: each wave issues
// exactly 3 async-load instructions per slab, so "s_wait_asynccnt 3" after
// issuing slab i+1 guarantees slab i has landed (one-deep pipeline).
__global__ __launch_bounds__(256) void gemm_f16_kernel(
    const _Float16* __restrict__ A, const _Float16* __restrict__ Bt,
    const float* __restrict__ bias, float* __restrict__ C,
    int M, int Nn, int K) {
  __shared__ _Float16 As[2][64][40];     // 32 valid halfs + pad
  __shared__ _Float16 Bs[2][128][40];
  const int tid = threadIdx.x, lane = tid & 31, wave = tid >> 5;
  const int ln = lane & 15, hf = lane >> 4;
  const int wm = (wave & 3) * 16, wn = (wave >> 2) * 64;
  const int bm = blockIdx.x * 64, bn = blockIdx.y * 128;
  const int ra = tid >> 2, koa = (tid & 3) * 8;   // A: 64 rows x 32 halfs
  const int rb = tid >> 1, kob = (tid & 1) * 16;  // B: 128 rows x 32 halfs
  const _Float16* Ar = A + (size_t)(bm + ra) * K + koa;
  const _Float16* Br = Bt + (size_t)(bn + rb) * K + kob;

  v8f acc[4] = {};
  // prologue: stage slab 0
  async_copy_b128(Ar, &As[0][ra][koa]);
  async_copy_b128(Br, &Bs[0][rb][kob]);
  async_copy_b128(Br + 8, &Bs[0][rb][kob + 8]);

  const int nslab = K / 32;
  for (int i = 0; i < nslab; ++i) {
    const int cur = i & 1;
    if (i + 1 < nslab) {
      const int kb = (i + 1) * 32;
      async_copy_b128(Ar + kb, &As[cur ^ 1][ra][koa]);
      async_copy_b128(Br + kb, &Bs[cur ^ 1][rb][kob]);
      async_copy_b128(Br + kb + 8, &Bs[cur ^ 1][rb][kob + 8]);
      WAIT_ASYNCCNT(3);   // slab i landed; slab i+1 still in flight
    } else {
      WAIT_ASYNCCNT(0);
    }
    __syncthreads();
    v16h a = fragA(&As[cur][wm + ln][0], 0, hf);
#pragma unroll
    for (int nt = 0; nt < 4; ++nt) {
      v16h b = fragB(&Bs[cur][wn + nt * 16 + ln][hf * 16]);
      acc[nt] = wmma32(a, b, acc[nt]);
    }
    __syncthreads();   // protect buffer reuse before next issue
  }
#pragma unroll
  for (int nt = 0; nt < 4; ++nt) {
    const int col = bn + wn + nt * 16 + ln;
    const float bv = bias[col];
#pragma unroll
    for (int j = 0; j < 8; ++j) {
      const int row = bm + wm + j + 8 * hf;
      C[(size_t)row * Nn + col] = acc[nt][j] + bv;
    }
  }
}

// ---------------------------------------------------------------- K2: depthwise conv3d pool + head-dim LN, one wave / output pos
__global__ __launch_bounds__(128) void pool_ln_kernel(
    const float* __restrict__ qkv,
    const float* __restrict__ wk, const float* __restrict__ wv,
    const float* __restrict__ lnkw, const float* __restrict__ lnkb,
    const float* __restrict__ lnvw, const float* __restrict__ lnvb,
    _Float16* __restrict__ k_h, _Float16* __restrict__ v_t) {
  const int gw = blockIdx.x * 4 + (threadIdx.x >> 5);
  if (gw >= BATCH * HEADS * NKV) return;
  const int lane = threadIdx.x & 31;
  const int bh = gw / NKV, pos = gw % NKV;
  const int b = bh / HEADS, h = bh % HEADS;
  const int ot = pos / 49, rr = pos % 49, oy = rr / 7, ox = rr % 7;
  float ak[2] = {0.f, 0.f}, av[2] = {0.f, 0.f};
#pragma unroll
  for (int c = 0; c < 2; ++c) {
    const int d = lane + 32 * c;
    for (int dt = 0; dt < 3; ++dt) {
      const int t = ot + dt - 1;
      if (t < 0 || t >= TT) continue;
      for (int dy = 0; dy < 3; ++dy) {
        const int y = 2 * oy + dy - 1;
        if (y < 0 || y >= HH) continue;
        for (int dx = 0; dx < 3; ++dx) {
          const int x = 2 * ox + dx - 1;
          if (x < 0 || x >= WW) continue;
          const size_t row = (size_t)b * NQ + t * (HH * WW) + y * WW + x;
          const float kin = qkv[row * QKVCOL + DIM     + h * HD + d];
          const float vin = qkv[row * QKVCOL + 2 * DIM + h * HD + d];
          const int widx = d * 27 + dt * 9 + dy * 3 + dx;
          ak[c] += kin * wk[widx];
          av[c] += vin * wv[widx];
        }
      }
    }
  }
  // LayerNorm over the 64 channels (2 per lane)
  const float mk = wave_sum(ak[0] + ak[1]) * (1.0f / HD);
  const float mv = wave_sum(av[0] + av[1]) * (1.0f / HD);
  float vk = (ak[0] - mk) * (ak[0] - mk) + (ak[1] - mk) * (ak[1] - mk);
  float vv = (av[0] - mv) * (av[0] - mv) + (av[1] - mv) * (av[1] - mv);
  const float rk = rsqrtf(wave_sum(vk) * (1.0f / HD) + EPSLN);
  const float rv = rsqrtf(wave_sum(vv) * (1.0f / HD) + EPSLN);
#pragma unroll
  for (int c = 0; c < 2; ++c) {
    const int d = lane + 32 * c;
    const float ok = (ak[c] - mk) * rk * lnkw[d] + lnkb[d];
    const float ov = (av[c] - mv) * rv * lnvw[d] + lnvb[d];
    k_h[((size_t)bh * NKP + pos) * HD + d] = (_Float16)ok;
    v_t[((size_t)bh * HD + d) * NKP2 + pos] = (_Float16)ov;
  }
}

// ---------------------------------------------------------------- K3: attention (one wave per (b,h,16-query tile))
__global__ __launch_bounds__(32) void attn_kernel(
    const float* __restrict__ qkv,
    const _Float16* __restrict__ k_h, const _Float16* __restrict__ v_t,
    const float* __restrict__ rph, const float* __restrict__ rpw,
    const float* __restrict__ rpt, _Float16* __restrict__ out_h) {
  __shared__ float    qf[16][HD];
  __shared__ _Float16 qh[16][HD];
  __shared__ float    ph[16][7], pw[16][7], pt[16][8];
  __shared__ float    sc[16][NKP];
  __shared__ _Float16 pb[16][NKP2];

  const int blk = blockIdx.x;
  const int qt = blk % (NQ / 16);       // 98 q-tiles
  const int bh = blk / (NQ / 16);
  const int b = bh / HEADS, h = bh % HEADS;
  const int lane = threadIdx.x, ln = lane & 15, hf = lane >> 4;
  const int q0 = qt * 16;

  // stage q tile (unscaled): fp32 for rel-pos/residual, f16 for WMMA A
  for (int i = lane; i < 16 * HD; i += 32) {
    const int r = i >> 6, d = i & 63;
    const float v = qkv[((size_t)b * NQ + q0 + r) * QKVCOL + h * HD + d];
    qf[r][d] = v;
    qh[r][d] = (_Float16)v;
  }
  __syncthreads();

  // rel-pos dot tables: 16 rows x (7 + 7 + 8) dots of length 64
  for (int i = lane; i < 16 * 22; i += 32) {
    const int r = i / 22, j = i % 22;
    const int n = q0 + r;
    const int t = n / (HH * WW), rem = n % (HH * WW), y = rem / WW, x = rem % WW;
    const float* R;
    if (j < 7)       R = rph + (size_t)(y - 2 * j + 12) * HD;
    else if (j < 14) R = rpw + (size_t)(x - 2 * (j - 7) + 12) * HD;
    else             R = rpt + (size_t)(t - (j - 14) + 7) * HD;
    float acc = 0.f;
    for (int d = 0; d < HD; ++d) acc += qf[r][d] * R[d];
    if (j < 7)       ph[r][j] = acc;
    else if (j < 14) pw[r][j - 7] = acc;
    else             pt[r][j - 14] = acc;
  }
  __syncthreads();

  // scores: q(16x64) @ k^T, 25 key tiles, K=64 -> 2 WMMA steps each
  const v16h a0 = fragA(&qh[ln][0], 0, hf);
  const v16h a1 = fragA(&qh[ln][0], 32, hf);
  const float scale = 0.125f;  // HD^-0.5
  const _Float16* kb = k_h + (size_t)bh * NKP * HD;
  for (int nt = 0; nt < NKP / 16; ++nt) {
    const _Float16* krow = kb + (size_t)(nt * 16 + ln) * HD;
    v8f c = {};
    c = wmma32(a0, fragB(krow + hf * 16), c);
    c = wmma32(a1, fragB(krow + 32 + hf * 16), c);
    const int col = nt * 16 + ln;
    int kt = 0, ky = 0, kx = 0;
    if (col < NKV) { kt = col / 49; const int r2 = col % 49; ky = r2 / 7; kx = r2 % 7; }
#pragma unroll
    for (int j = 0; j < 8; ++j) {
      const int r = j + 8 * hf;
      float sv;
      if (col < NKV) sv = c[j] * scale + ph[r][ky] + pw[r][kx] + pt[r][kt];
      else           sv = -1e30f;
      sc[r][col] = sv;
    }
  }
  __syncthreads();

  // softmax rows -> f16 probabilities (padded cols zeroed)
  for (int r = 0; r < 16; ++r) {
    float m = -1e30f;
    for (int c2 = lane; c2 < NKV; c2 += 32) m = fmaxf(m, sc[r][c2]);
    m = wave_max(m);
    float s = 0.f;
    for (int c2 = lane; c2 < NKV; c2 += 32) {
      const float e = __expf(sc[r][c2] - m);
      sc[r][c2] = e;
      s += e;
    }
    s = wave_sum(s);
    const float inv = 1.0f / s;
    for (int c2 = lane; c2 < NKP2; c2 += 32)
      pb[r][c2] = (_Float16)((c2 < NKV) ? sc[r][c2] * inv : 0.f);
  }
  __syncthreads();

  // out = P(16x416) @ v(416x64), 13 K-steps x 4 d-tiles
  v8f o[4] = {};
  const _Float16* vb = v_t + (size_t)bh * HD * NKP2;
  for (int ks = 0; ks < NKP2 / 32; ++ks) {
    const v16h a = fragA(&pb[ln][0], ks * 32, hf);
#pragma unroll
    for (int nt = 0; nt < 4; ++nt) {
      const _Float16* vrow = vb + (size_t)(nt * 16 + ln) * NKP2 + ks * 32 + hf * 16;
      o[nt] = wmma32(a, fragB(vrow), o[nt]);
    }
  }
  // residual (+q) and write f16 panel for the projection GEMM
#pragma unroll
  for (int nt = 0; nt < 4; ++nt) {
    const int d = nt * 16 + ln;
#pragma unroll
    for (int j = 0; j < 8; ++j) {
      const int r = j + 8 * hf;
      const float val = o[nt][j] + qf[r][d];
      out_h[((size_t)b * NQ + q0 + r) * DIM + h * HD + d] = (_Float16)val;
    }
  }
}

// ---------------------------------------------------------------- launcher
static inline size_t alignup(size_t x) { return (x + 255) & ~(size_t)255; }

extern "C" void kernel_launch(void* const* d_in, const int* in_sizes, int n_in,
                              void* d_out, int out_size, void* d_ws, size_t ws_size,
                              hipStream_t stream) {
  (void)in_sizes; (void)n_in; (void)out_size; (void)ws_size;
  const float* x      = (const float*)d_in[0];
  const float* ln_w   = (const float*)d_in[1];
  const float* ln_b   = (const float*)d_in[2];
  const float* qkv_w  = (const float*)d_in[3];
  const float* qkv_b  = (const float*)d_in[4];
  const float* proj_w = (const float*)d_in[5];
  const float* proj_b = (const float*)d_in[6];
  const float* poolk  = (const float*)d_in[7];
  const float* poolv  = (const float*)d_in[8];
  const float* lnk_w  = (const float*)d_in[9];
  const float* lnk_b  = (const float*)d_in[10];
  const float* lnv_w  = (const float*)d_in[11];
  const float* lnv_b  = (const float*)d_in[12];
  const float* rph    = (const float*)d_in[13];
  const float* rpw    = (const float*)d_in[14];
  const float* rpt    = (const float*)d_in[15];

  char* ws = (char*)d_ws;
  size_t off = 0;
  _Float16* xn_h    = (_Float16*)(ws + off); off += alignup((size_t)ROWS * DIM * 2);
  _Float16* qkvw_t  = (_Float16*)(ws + off); off += alignup((size_t)QKVCOL * DIM * 2);
  _Float16* projw_t = (_Float16*)(ws + off); off += alignup((size_t)DIM * DIM * 2);
  float*    qkvbuf  = (float*)   (ws + off); off += alignup((size_t)ROWS * QKVCOL * 4);
  _Float16* k_h     = (_Float16*)(ws + off); off += alignup((size_t)BATCH * HEADS * NKP * HD * 2);
  _Float16* v_t     = (_Float16*)(ws + off); off += alignup((size_t)BATCH * HEADS * HD * NKP2 * 2);
  _Float16* out_h   = (_Float16*)(ws + off); off += alignup((size_t)ROWS * DIM * 2);

  // 0) layernorm(x) -> f16
  ln_cast_kernel<<<ROWS, 256, 0, stream>>>(x, ln_w, ln_b, xn_h);
  // 0b) weights -> f16 N-major
  {
    int n = DIM * QKVCOL;
    tcast_kernel<<<(n + 255) / 256, 256, 0, stream>>>(qkv_w, qkvw_t, DIM, QKVCOL);
    n = DIM * DIM;
    tcast_kernel<<<(n + 255) / 256, 256, 0, stream>>>(proj_w, projw_t, DIM, DIM);
  }
  // zero padded k/v panels
  {
    int nk = BATCH * HEADS * NKP * HD;
    zero_h_kernel<<<(nk + 255) / 256, 256, 0, stream>>>(k_h, nk);
    int nv = BATCH * HEADS * HD * NKP2;
    zero_h_kernel<<<(nv + 255) / 256, 256, 0, stream>>>(v_t, nv);
  }
  // 1) QKV GEMM: (6272x768) @ (768x2304) + b
  {
    dim3 grid(ROWS / 64, QKVCOL / 128);
    gemm_f16_kernel<<<grid, 256, 0, stream>>>(xn_h, qkvw_t, qkv_b, qkvbuf,
                                              ROWS, QKVCOL, DIM);
  }
  // 2) conv pool + LN of k/v
  {
    int waves = BATCH * HEADS * NKV;  // 18816, /4 waves per block
    pool_ln_kernel<<<waves / 4, 128, 0, stream>>>(qkvbuf, poolk, poolv,
                                                  lnk_w, lnk_b, lnv_w, lnv_b,
                                                  k_h, v_t);
  }
  // 3) attention
  attn_kernel<<<BATCH * HEADS * (NQ / 16), 32, 0, stream>>>(
      qkvbuf, k_h, v_t, rph, rpw, rpt, out_h);
  // 4) projection GEMM -> d_out (fp32)
  {
    dim3 grid(ROWS / 64, DIM / 128);
    gemm_f16_kernel<<<grid, 256, 0, stream>>>(out_h, projw_t, proj_b,
                                              (float*)d_out, ROWS, DIM, DIM);
  }
}
